// MultiHeadAttention_45990509805663
// MI455X (gfx1250) — matrix operand order repaired and checked by reading, not compiled, not run on hardware
//
#include <hip/hip_runtime.h>
#include <hip/hip_bf16.h>

// ---------------------------------------------------------------------------
// MultiHeadAttention + residual + LayerNorm for MI455X (gfx1250, wave32).
// All GEMMs run as NT bf16 WMMA (v_wmma_f32_16x16x32_bf16) with fp32 accum.
// Streaming attn surface (537 MB) uses non-temporal loads/stores to keep the
// 192 MB L2 for the reusable tensors (QH/KH/VT/CTX/weights).
// Fixed problem size: B=4, L=2048, D=1024, H=8, DK=DV=128.
// ---------------------------------------------------------------------------

typedef __bf16 bf16_t;
typedef __attribute__((ext_vector_type(16))) __bf16 v16bf;
typedef __attribute__((ext_vector_type(8)))  __bf16 v8bf;
typedef __attribute__((ext_vector_type(8)))  float  v8f;

#define B_  4
#define L_  2048
#define D_  1024
#define H_  8
#define DK_ 128

#define MODE_F32   0   // C fp32 row-major (+ per-z offset), val *= scale
#define MODE_HEAD  1   // C bf16 scattered to [H,B,L,DK]
#define MODE_HEADT 2   // C bf16 scattered to [H,B,DK,L]   (V transposed)
#define MODE_CTX   3   // C bf16 scattered to [B,L,H*DV]   (per-z h,b)

// ---------------------------------------------------------------------------
// fp32 -> bf16 elementwise convert
// ---------------------------------------------------------------------------
__global__ __launch_bounds__(256) void cvt_bf16_kernel(const float* __restrict__ in,
                                                       bf16_t* __restrict__ out,
                                                       unsigned int n) {
    unsigned int i = blockIdx.x * blockDim.x + threadIdx.x;
    unsigned int stride = gridDim.x * blockDim.x;
    for (; i < n; i += stride) out[i] = (bf16_t)in[i];
}

// ---------------------------------------------------------------------------
// WMMA fragment loaders (wave32).
// A (16x32 bf16): lanes 0-15 -> row M=lane, elems 0..7 = K[k..k+7],
//                 elems 8..15 = K[k+16..k+23]; lanes 16-31 same row, K +8.
// B (32x16 bf16): lanes 0-15 -> col N=lane, elems 0..7 = K[k..k+7],
//                 elems 8..15 = K[k+16..k+23]; lanes 16-31 same col, K +8.
// ---------------------------------------------------------------------------
__device__ __forceinline__ v16bf load_a_bf(const bf16_t* A, int lda, int r0, int k, int lane) {
    const int rr = r0 + (lane & 15);
    const int ko = k + ((lane >> 4) << 3);
    const bf16_t* p = A + (size_t)rr * lda + ko;
    v8bf lo = *(const v8bf*)(p);
    v8bf hi = *(const v8bf*)(p + 16);
    v16bf f;
#pragma unroll
    for (int i = 0; i < 8; i++) { f[i] = lo[i]; f[i + 8] = hi[i]; }
    return f;
}

// fp32 A with on-the-fly bf16 convert; non-temporal (streamed attn surface)
__device__ __forceinline__ v16bf load_a_f32_nt(const float* A, int lda, int r0, int k, int lane) {
    const int rr = r0 + (lane & 15);
    const int ko = k + ((lane >> 4) << 3);
    const float* p = A + (size_t)rr * lda + ko;
    const v8f lo = __builtin_nontemporal_load((const v8f*)p);
    const v8f hi = __builtin_nontemporal_load((const v8f*)(p + 16));
    v16bf f;
#pragma unroll
    for (int i = 0; i < 8; i++) { f[i] = (bf16_t)lo[i]; f[i + 8] = (bf16_t)hi[i]; }
    return f;
}

__device__ __forceinline__ v16bf load_b_bf(const bf16_t* Bm, int ldb, int c0, int k, int lane) {
    const int cc = c0 + (lane & 15);
    const int ko = k + ((lane >> 4) << 3);
    const bf16_t* p = Bm + (size_t)cc * ldb + ko;
    v8bf lo = *(const v8bf*)(p);
    v8bf hi = *(const v8bf*)(p + 16);
    v16bf f;
#pragma unroll
    for (int i = 0; i < 8; i++) { f[i] = lo[i]; f[i + 8] = hi[i]; }
    return f;
}

// ---------------------------------------------------------------------------
// NT GEMM: C[M,N] = scale * A[M,K] @ B[N,K]^T  (per blockIdx.z slice).
// 256 threads = 8 waves; each wave computes one 64x64 tile (4x4 WMMA frags).
// Requires M%64==0, N%64==0, K%32==0, gridDim.x*8 == (M/64)*(N/64).
// NTC: non-temporal C stores (for the streaming attn surface).
// ---------------------------------------------------------------------------
template<int MODE, bool AF32, bool NTC>
__global__ __launch_bounds__(256) void gemm_nt_kernel(
    const void* __restrict__ Ap, const bf16_t* __restrict__ Bp, void* __restrict__ Cp,
    int Mdim, int Ndim, int Kdim, int lda, int ldb,
    long long aZ, long long bZ, long long cZ, float scale)
{
    const int lane   = threadIdx.x & 31;
    const int wave   = threadIdx.x >> 5;
    const int z      = blockIdx.z;
    const int tilesN = Ndim >> 6;
    const int tileId = blockIdx.x * 8 + wave;
    const int tm     = tileId / tilesN;
    const int tn     = tileId - tm * tilesN;
    const int row0   = tm << 6;
    const int col0   = tn << 6;

    const float*  Af = nullptr;
    const bf16_t* Ab = nullptr;
    if constexpr (AF32) Af = (const float*)Ap + (size_t)z * (size_t)aZ;
    else                Ab = (const bf16_t*)Ap + (size_t)z * (size_t)aZ;
    const bf16_t* Bb = Bp + (size_t)z * (size_t)bZ;

    v8f acc[4][4];
#pragma unroll
    for (int i = 0; i < 4; i++)
#pragma unroll
        for (int j = 0; j < 4; j++)
#pragma unroll
            for (int e = 0; e < 8; e++) acc[i][j][e] = 0.0f;

    for (int k = 0; k < Kdim; k += 32) {
        // speculative WGP-scope prefetch of next K block (locality 3 -> near caches)
        if constexpr (AF32)
            __builtin_prefetch(Af + (size_t)(row0 + (lane & 15)) * lda + k + 32, 0, 3);
        else
            __builtin_prefetch(Ab + (size_t)(row0 + (lane & 15)) * lda + k + 32, 0, 3);
        __builtin_prefetch(Bb + (size_t)(col0 + (lane & 15)) * ldb + k + 32, 0, 3);

        v16bf afr[4], bfr[4];
#pragma unroll
        for (int fi = 0; fi < 4; fi++) {
            if constexpr (AF32) afr[fi] = load_a_f32_nt(Af, lda, row0 + fi * 16, k, lane);
            else                afr[fi] = load_a_bf   (Ab, lda, row0 + fi * 16, k, lane);
        }
#pragma unroll
        for (int fj = 0; fj < 4; fj++)
            bfr[fj] = load_b_bf(Bb, ldb, col0 + fj * 16, k, lane);

#pragma unroll
        for (int fi = 0; fi < 4; fi++)
#pragma unroll
            for (int fj = 0; fj < 4; fj++)
                acc[fi][fj] = __builtin_amdgcn_wmma_f32_16x16x32_bf16(
                    false, afr[fi], false, bfr[fj], (short)0, acc[fi][fj], false, false);
    }

    // C layout: lane 0-15 -> N=lane, M = frag_row + i; lane 16-31 -> M += 8
    const int nlane = lane & 15;
    const int msub  = (lane >> 4) << 3;
#pragma unroll
    for (int fi = 0; fi < 4; fi++) {
#pragma unroll
        for (int fj = 0; fj < 4; fj++) {
            const int n  = col0 + fj * 16 + nlane;
            const int mb = row0 + fi * 16 + msub;
            v8f c = acc[fi][fj];
#pragma unroll
            for (int i = 0; i < 8; i++) {
                const int m = mb + i;
                const float val = c[i] * scale;
                if constexpr (MODE == MODE_F32) {
                    float* dst = (float*)Cp + (size_t)z * (size_t)cZ + (size_t)m * Ndim + n;
                    if constexpr (NTC) __builtin_nontemporal_store(val, dst);
                    else               *dst = val;
                } else if constexpr (MODE == MODE_HEAD) {
                    const int h = n >> 7, d = n & 127, b = m >> 11, l = m & 2047;
                    ((bf16_t*)Cp)[((size_t)(h * B_ + b) * L_ + l) * DK_ + d] = (bf16_t)val;
                } else if constexpr (MODE == MODE_HEADT) {
                    const int h = n >> 7, d = n & 127, b = m >> 11, l = m & 2047;
                    ((bf16_t*)Cp)[((size_t)(h * B_ + b) * DK_ + d) * L_ + l] = (bf16_t)val;
                } else {  // MODE_CTX: z = h*B + b, m = l, n = dv
                    const int h = z >> 2, b = z & 3;
                    ((bf16_t*)Cp)[(size_t)(b * L_ + m) * D_ + h * DK_ + n] = (bf16_t)val;
                }
            }
        }
    }
}

// ---------------------------------------------------------------------------
// masked softmax over one attn row; grid = (L, H*B), 256 threads, 8 elems each.
// attn is a streamed 537 MB surface -> non-temporal row load/store.
// ---------------------------------------------------------------------------
__global__ __launch_bounds__(256) void softmax_kernel(float* __restrict__ attn,
                                                      const int* __restrict__ mask) {
    const int qrow = blockIdx.x;
    const int z    = blockIdx.y;
    const int b    = z & (B_ - 1);
    float* row = attn + ((size_t)z * L_ + qrow) * L_;
    const int* mrow = mask + (size_t)b * L_;
    const int tid = threadIdx.x;
    __shared__ float red[256];

    float vals[8];
    float mx = -__builtin_inff();
#pragma unroll
    for (int j = 0; j < 8; j++) {
        const int col = tid + j * 256;
        const float rv = __builtin_nontemporal_load(row + col);
        const float s = (mrow[col] != 0) ? rv : -__builtin_inff();
        vals[j] = s;
        mx = fmaxf(mx, s);
    }
    red[tid] = mx; __syncthreads();
    for (int s = 128; s > 0; s >>= 1) {
        if (tid < s) red[tid] = fmaxf(red[tid], red[tid + s]);
        __syncthreads();
    }
    mx = red[0]; __syncthreads();

    float sum = 0.0f;
#pragma unroll
    for (int j = 0; j < 8; j++) { vals[j] = __expf(vals[j] - mx); sum += vals[j]; }
    red[tid] = sum; __syncthreads();
    for (int s = 128; s > 0; s >>= 1) {
        if (tid < s) red[tid] += red[tid + s];
        __syncthreads();
    }
    const float inv = 1.0f / red[0];
#pragma unroll
    for (int j = 0; j < 8; j++)
        __builtin_nontemporal_store(vals[j] * inv, row + tid + j * 256);
}

// ---------------------------------------------------------------------------
// y = LayerNorm(oproj + residual) * gamma + beta ; one block per row of 1024
// ---------------------------------------------------------------------------
__global__ __launch_bounds__(256) void ln_kernel(const float* __restrict__ oproj,
                                                 const float* __restrict__ residual,
                                                 const float* __restrict__ gamma,
                                                 const float* __restrict__ beta,
                                                 float* __restrict__ y) {
    const int rowi = blockIdx.x;
    const int tid  = threadIdx.x;
    const float* xo = oproj + (size_t)rowi * D_;
    const float* rr = residual + (size_t)rowi * D_;
    __shared__ float s1[256];
    __shared__ float s2[256];

    float x[4], s = 0.0f, sq = 0.0f;
#pragma unroll
    for (int j = 0; j < 4; j++) {
        const int col = tid + j * 256;
        const float v = xo[col] + rr[col];
        x[j] = v; s += v; sq += v * v;
    }
    s1[tid] = s; s2[tid] = sq; __syncthreads();
    for (int st = 128; st > 0; st >>= 1) {
        if (tid < st) { s1[tid] += s1[tid + st]; s2[tid] += s2[tid + st]; }
        __syncthreads();
    }
    const float mu  = s1[0] * (1.0f / D_);
    const float var = s2[0] * (1.0f / D_) - mu * mu;
    const float inv = rsqrtf(var + 1e-5f);
#pragma unroll
    for (int j = 0; j < 4; j++) {
        const int col = tid + j * 256;
        const float o = (x[j] - mu) * inv * gamma[col] + beta[col];
        __builtin_nontemporal_store(o, y + (size_t)rowi * D_ + col);
    }
}

// ---------------------------------------------------------------------------
// Host-side orchestration (all on `stream`; graph-capture safe).
// Workspace layout (bf16 elements unless noted), peak ~120 MB:
//   [0, 4M)    : Wq,Wk,Wv,Wo bf16            (4 x 1M)
//   [4M, 28M)  : q,k,v bf16 (8M each)        -- reused later as fp32 OPROJ
//   [28M, 52M) : QH [H,B,L,DK], KH, VT [H,B,DK,L]  (8M each)
//   [52M, 60M) : CTX [B,L,H*DV]
// ---------------------------------------------------------------------------
extern "C" void kernel_launch(void* const* d_in, const int* in_sizes, int n_in,
                              void* d_out, int out_size, void* d_ws, size_t ws_size,
                              hipStream_t stream) {
    (void)in_sizes; (void)n_in; (void)out_size; (void)ws_size;

    const float* q    = (const float*)d_in[0];
    const float* k    = (const float*)d_in[1];
    const float* v    = (const float*)d_in[2];
    const int*   mask = (const int*)  d_in[3];
    const float* Wq   = (const float*)d_in[4];
    const float* Wk   = (const float*)d_in[5];
    const float* Wv   = (const float*)d_in[6];
    const float* Wo   = (const float*)d_in[7];
    const float* gamma= (const float*)d_in[8];
    const float* beta = (const float*)d_in[9];

    float* y    = (float*)d_out;
    float* attn = y + (size_t)B_ * L_ * D_;   // 32 MB offset; [H*B, L, L] fp32

    bf16_t* Wq_bf = (bf16_t*)d_ws;
    bf16_t* Wk_bf = Wq_bf + (1u << 20);
    bf16_t* Wv_bf = Wk_bf + (1u << 20);
    bf16_t* Wo_bf = Wv_bf + (1u << 20);
    bf16_t* Xq    = Wo_bf + (1u << 20);
    bf16_t* Xk    = Xq + (8u << 20);
    bf16_t* Xv    = Xk + (8u << 20);
    bf16_t* QH    = Xv + (8u << 20);
    bf16_t* KH    = QH + (8u << 20);
    bf16_t* VT    = KH + (8u << 20);
    bf16_t* CTX   = VT + (8u << 20);
    float*  OPROJ = (float*)Xq;               // reuse 48MB region (needs 32MB)

    const unsigned int NX = 8u << 20;   // 8192*1024
    const unsigned int NW = 1u << 20;   // 1024*1024

    // 1) fp32 -> bf16 conversions
    cvt_bf16_kernel<<<2048, 256, 0, stream>>>(q,  Xq, NX);
    cvt_bf16_kernel<<<2048, 256, 0, stream>>>(k,  Xk, NX);
    cvt_bf16_kernel<<<2048, 256, 0, stream>>>(v,  Xv, NX);
    cvt_bf16_kernel<<<512,  256, 0, stream>>>(Wq, Wq_bf, NW);
    cvt_bf16_kernel<<<512,  256, 0, stream>>>(Wk, Wk_bf, NW);
    cvt_bf16_kernel<<<512,  256, 0, stream>>>(Wv, Wv_bf, NW);
    cvt_bf16_kernel<<<512,  256, 0, stream>>>(Wo, Wo_bf, NW);

    // 2) QKV projections: [8192,1024] @ [1024,1024]^T, head-split outputs
    //    tiles = 128*16 = 2048 -> grid.x = 256
    gemm_nt_kernel<MODE_HEAD,  false, false><<<dim3(256, 1, 1), 256, 0, stream>>>(
        Xq, Wq_bf, QH, 8192, 1024, 1024, 1024, 1024, 0, 0, 0, 1.0f);
    gemm_nt_kernel<MODE_HEAD,  false, false><<<dim3(256, 1, 1), 256, 0, stream>>>(
        Xk, Wk_bf, KH, 8192, 1024, 1024, 1024, 1024, 0, 0, 0, 1.0f);
    gemm_nt_kernel<MODE_HEADT, false, false><<<dim3(256, 1, 1), 256, 0, stream>>>(
        Xv, Wv_bf, VT, 8192, 1024, 1024, 1024, 1024, 0, 0, 0, 1.0f);

    // 3) scores = (qh @ kh^T)/sqrt(DK) per z=(h*B+b): [2048,128]x[2048,128]^T
    //    tiles = 32*32 = 1024 -> grid.x = 128, grid.z = 32 ; NT stores (537 MB)
    gemm_nt_kernel<MODE_F32, false, true><<<dim3(128, 1, 32), 256, 0, stream>>>(
        QH, KH, attn, 2048, 2048, 128, 128, 128,
        (long long)L_ * DK_, (long long)L_ * DK_, (long long)L_ * L_,
        0.08838834764831845f);

    // 4) masked softmax rows (NT read/write of attn, in-place)
    softmax_kernel<<<dim3(L_, H_ * B_), 256, 0, stream>>>(attn, mask);

    // 5) ctx = attn @ vh  (A fp32 NT-loaded + converted on the fly; B = VT [DK, L])
    //    tiles = 32*2 = 64 -> grid.x = 8, grid.z = 32
    gemm_nt_kernel<MODE_CTX, true, false><<<dim3(8, 1, 32), 256, 0, stream>>>(
        attn, VT, CTX, 2048, 128, 2048, 2048, 2048,
        (long long)L_ * L_, (long long)DK_ * L_, 0, 1.0f);

    // 6) output projection: [8192,1024] @ Wo[1024,1024]^T -> OPROJ fp32 (L2-resident)
    gemm_nt_kernel<MODE_F32, false, false><<<dim3(256, 1, 1), 256, 0, stream>>>(
        CTX, Wo_bf, OPROJ, 8192, 1024, 1024, 1024, 1024, 0, 0, 0, 1.0f);

    // 7) residual + LayerNorm -> y (streamed out)
    ln_kernel<<<8192, 256, 0, stream>>>(OPROJ, q, gamma, beta, y);
}
